// RotDeltaAttention_73452530696275
// MI455X (gfx1250) — matrix operand order repaired and checked by reading, not compiled, not run on hardware
//
#include <hip/hip_runtime.h>
#include <hip/hip_bf16.h>

// ---------------------------------------------------------------------------
// RotDeltaAttention for MI455X (gfx1250, wave32, WMMA)
//   D=1024, H=16, N=8, NSKEW=28, B=4, T=4096
// Pipeline:
//   1. cvt x -> bf16
//   2. pack weights (transposed, zero-padded, bf16): Wt[896][1024], Wot[1024][128]
//   3. proj GEMM  [16384x1024]@[1024x896] via v_wmma_f32_16x16x32_bf16,
//      NT=4 register blocking (1 A frag reused across 4 B frags)
//   4. expm of 8x8 skews (per-thread, register-resident Taylor + squaring)
//   5. k-normalize + beta sigmoid
//   6. sequential scan: 64 chains, one wave32 each, state in LDS
//   7. out GEMM [16384x128]@[128x1024] via WMMA -> d_out (f32)
// ---------------------------------------------------------------------------

#define BB 4
#define TT 4096
#define DD 1024
#define HH 16
#define NN 8
#define NSKEW 28
#define BT (BB * TT)          // 16384
#define NCOLS 848             // 448 skew + 128 k + 128 v + 128 q + 16 beta
#define PCOLS 896             // padded to a multiple of 64 for NT=4 blocking
#define COL_K 448
#define COL_V 576
#define COL_Q 704
#define COL_B 832
#define NT 4                  // N tiles per wave (register blocking)

typedef __attribute__((ext_vector_type(16))) __bf16 v16bf;
typedef __attribute__((ext_vector_type(8)))  float  v8f;

// ---------------------------------------------------------------------------
// f32 -> bf16 conversion, 4 elements/thread (memory bound)
// ---------------------------------------------------------------------------
__global__ __launch_bounds__(256) void cvt_f32_bf16_x4(
    const float* __restrict__ in, __bf16* __restrict__ out, size_t n4)
{
    size_t i = (size_t)blockIdx.x * blockDim.x + threadIdx.x;
    if (i >= n4) return;
    size_t b = i * 4;
    #pragma unroll
    for (int j = 0; j < 4; ++j) out[b + j] = (__bf16)in[b + j];
}

// ---------------------------------------------------------------------------
// Pack the five projection weights into Wt[col][k] (transposed, bf16),
// zero-padding cols [848, 896).
// ---------------------------------------------------------------------------
__global__ __launch_bounds__(256) void pack_weights(
    const float* __restrict__ Ws, const float* __restrict__ Wk,
    const float* __restrict__ Wv, const float* __restrict__ Wq,
    const float* __restrict__ Wb, __bf16* __restrict__ Wt)
{
    int idx = blockIdx.x * blockDim.x + threadIdx.x;   // col*1024 + k
    if (idx >= PCOLS * DD) return;
    int col = idx >> 10;
    int k   = idx & (DD - 1);
    float v;
    if      (col < COL_K)  v = Ws[(size_t)k * 448 + col];
    else if (col < COL_V)  v = Wk[(size_t)k * 128 + (col - COL_K)];
    else if (col < COL_Q)  v = Wv[(size_t)k * 128 + (col - COL_V)];
    else if (col < COL_B)  v = Wq[(size_t)k * 128 + (col - COL_Q)];
    else if (col < NCOLS)  v = Wb[(size_t)k * 16  + (col - COL_B)];
    else                   v = 0.f;                     // pad
    Wt[idx] = (__bf16)v;
}

// W_o [128,1024] f32 -> Wot[1024][128] bf16
__global__ __launch_bounds__(256) void pack_wo(
    const float* __restrict__ Wo, __bf16* __restrict__ Wot)
{
    int idx = blockIdx.x * blockDim.x + threadIdx.x;   // col*128 + k
    if (idx >= 1024 * 128) return;
    int col = idx >> 7;
    int k   = idx & 127;
    Wot[idx] = (__bf16)Wo[(size_t)k * 1024 + col];
}

// ---------------------------------------------------------------------------
// bf16 WMMA GEMM: C[M,N] = A[M,K] @ Bt[N,K]^T, fp32 accumulate.
// One wave per 16x64 C strip (NT=4 accumulators); 8 waves/block along M;
// grid = (N/64, M/(16*8)).  Requires M%16==0, N%64==0, K%32==0.
// A fragment (16-bit A layout): lane<16 holds row m0+lane, K {0..7,16..23};
// lane>=16 same rows, K {8..15,24..31} -> two 16B contiguous loads, reused
// across 4 B fragments.  B fragment: lane<16 col n0+lane K {0..7,16..23};
// lane>=16 K {8..15,24..31} -> two 16B loads (Bt stored [col][K]).
// ---------------------------------------------------------------------------
__global__ __launch_bounds__(256) void gemm_bf16_wmma(
    const __bf16* __restrict__ A, const __bf16* __restrict__ Bt,
    float* __restrict__ C, int M, int Nc, int K)
{
    const int lane = threadIdx.x & 31;
    const int wave = threadIdx.x >> 5;
    const int hs   = lane >> 4;
    const int l16  = lane & 15;

    const int n0 = blockIdx.x * (16 * NT);
    const int m0 = (blockIdx.y * 8 + wave) * 16;
    if (m0 >= M) return;

    const __bf16* arow = A  + (size_t)(m0 + l16) * K;
    const __bf16* brow = Bt + (size_t)(n0 + l16) * K;

    v8f acc[NT] = {};
    for (int k0 = 0; k0 < K; k0 += 32) {
        v16bf a;
        const __bf16* ap = arow + k0 + hs * 8;
        __builtin_prefetch(ap + 256, 0, 0);            // global_prefetch_b8
        #pragma unroll
        for (int i = 0; i < 8; ++i)  a[i]     = ap[i];
        #pragma unroll
        for (int i = 0; i < 8; ++i)  a[8 + i] = ap[16 + i];
        #pragma unroll
        for (int nt = 0; nt < NT; ++nt) {
            v16bf b;
            const __bf16* bp = brow + (size_t)nt * 16 * K + k0 + hs * 8;
            #pragma unroll
            for (int i = 0; i < 8; ++i) { b[i] = bp[i]; b[8 + i] = bp[16 + i]; }
            acc[nt] = __builtin_amdgcn_wmma_f32_16x16x32_bf16(
                false, a, false, b, (short)0, acc[nt], false, false);
        }
    }

    // D layout: VGPR r -> (M = hs*8 + r, N = l16)
    #pragma unroll
    for (int nt = 0; nt < NT; ++nt) {
        float* crow = C + (size_t)(m0 + hs * 8) * Nc + n0 + nt * 16 + l16;
        #pragma unroll
        for (int r = 0; r < 8; ++r) crow[(size_t)r * Nc] = acc[nt][r];
    }
}

// ---------------------------------------------------------------------------
// expm of 8x8 skew-symmetric matrices. One thread per (b,t,h).
// A = 0.5*tanh(skew_flat)/4 (scaled for 2 squarings); 10-term Taylor;
// exp(A0) = (exp(A0/4))^4. Fully unrolled -> register resident.
// ---------------------------------------------------------------------------
__global__ __launch_bounds__(128) void expm8_kernel(
    const float* __restrict__ proj, float* __restrict__ Obuf, int total)
{
    int tid = blockIdx.x * blockDim.x + threadIdx.x;
    if (tid >= total) return;
    int bt = tid >> 4, h = tid & 15;
    const float* sp = proj + (size_t)bt * PCOLS + h * NSKEW;

    float A[8][8];
    #pragma unroll
    for (int i = 0; i < 8; ++i)
        #pragma unroll
        for (int j = 0; j < 8; ++j) A[i][j] = 0.f;
    {
        int c = 0;
        #pragma unroll
        for (int i = 0; i < 8; ++i)
            #pragma unroll
            for (int j = i + 1; j < 8; ++j) {
                float s = 0.125f * tanhf(sp[c]);   // 0.5 * tanh * (1/4)
                A[i][j] = s; A[j][i] = -s; ++c;
            }
    }

    float R[8][8], Tm[8][8];
    #pragma unroll
    for (int i = 0; i < 8; ++i)
        #pragma unroll
        for (int j = 0; j < 8; ++j) {
            Tm[i][j] = A[i][j];
            R[i][j]  = A[i][j] + (i == j ? 1.f : 0.f);
        }

    for (int it = 2; it <= 10; ++it) {          // Taylor terms
        float inv = 1.f / (float)it;
        #pragma unroll
        for (int i = 0; i < 8; ++i) {
            float row[8];
            #pragma unroll
            for (int j = 0; j < 8; ++j) {
                float s = 0.f;
                #pragma unroll
                for (int k = 0; k < 8; ++k) s = fmaf(Tm[i][k], A[k][j], s);
                row[j] = s * inv;
            }
            #pragma unroll
            for (int j = 0; j < 8; ++j) { Tm[i][j] = row[j]; R[i][j] += row[j]; }
        }
    }

    #pragma unroll
    for (int sq = 0; sq < 2; ++sq) {            // two squarings
        float Q[8][8];
        #pragma unroll
        for (int i = 0; i < 8; ++i)
            #pragma unroll
            for (int j = 0; j < 8; ++j) {
                float s = 0.f;
                #pragma unroll
                for (int k = 0; k < 8; ++k) s = fmaf(R[i][k], R[k][j], s);
                Q[i][j] = s;
            }
        #pragma unroll
        for (int i = 0; i < 8; ++i)
            #pragma unroll
            for (int j = 0; j < 8; ++j) R[i][j] = Q[i][j];
    }

    float* op = Obuf + (size_t)tid * 64;
    #pragma unroll
    for (int i = 0; i < 8; ++i)
        #pragma unroll
        for (int j = 0; j < 8; ++j) op[i * 8 + j] = R[i][j];
}

// ---------------------------------------------------------------------------
// k normalization + beta sigmoid. One thread per (b,t,h).
// ---------------------------------------------------------------------------
__global__ __launch_bounds__(256) void normbeta_kernel(
    const float* __restrict__ proj, const float* __restrict__ b_beta,
    float* __restrict__ knorm, float* __restrict__ betab, int total)
{
    int tid = blockIdx.x * blockDim.x + threadIdx.x;
    if (tid >= total) return;
    int bt = tid >> 4, h = tid & 15;
    const float* kp = proj + (size_t)bt * PCOLS + COL_K + h * 8;
    float kv[8], s = 0.f;
    #pragma unroll
    for (int i = 0; i < 8; ++i) { kv[i] = kp[i]; s = fmaf(kv[i], kv[i], s); }
    float inv = 1.f / fmaxf(sqrtf(s), 1e-6f);
    float* ko = knorm + (size_t)bt * (HH * 8) + h * 8;
    #pragma unroll
    for (int i = 0; i < 8; ++i) ko[i] = kv[i] * inv;
    float z = proj[(size_t)bt * PCOLS + COL_B + h] + b_beta[h];
    betab[(size_t)bt * HH + h] = 1.f / (1.f + expf(-z));
}

// ---------------------------------------------------------------------------
// Sequential scan: one wave32 per (b,h) chain, 8x8 state in LDS.
// Single-wave workgroup -> barrier ops degrade to S_NOP + DS waits.
// ---------------------------------------------------------------------------
__global__ __launch_bounds__(32) void scan_kernel(
    const float* __restrict__ Obuf,   // [BT*H, 64]
    const float* __restrict__ knorm,  // [BT, 128]
    const float* __restrict__ proj,   // [BT, 896]
    const float* __restrict__ betab,  // [BT, 16]
    __bf16* __restrict__ o_bf)        // [BT, 128]
{
    const int h = blockIdx.x & 15;
    const int b = blockIdx.x >> 4;
    const int lane = threadIdx.x;

    __shared__ float c_s[64], t_s[64], r_s[64], O_s[64];
    __shared__ float k_s[8], v_s[8], q_s[8], kc_s[8];
    __shared__ float beta_s;

    c_s[lane] = 0.f; c_s[lane + 32] = 0.f;
    __syncthreads();

    const int e0 = lane, e1 = lane + 32;
    const int i0 = e0 >> 3, j0 = e0 & 7;
    const int i1 = e1 >> 3, j1 = e1 & 7;

    for (int t = 0; t < TT; ++t) {
        const size_t bt = (size_t)b * TT + t;
        const float* op = Obuf + (bt * HH + h) * 64;
        O_s[e0] = op[e0]; O_s[e1] = op[e1];
        if (lane < 8) {
            k_s[lane] = knorm[bt * 128 + h * 8 + lane];
            v_s[lane] = proj[bt * PCOLS + COL_V + h * 8 + lane];
            q_s[lane] = proj[bt * PCOLS + COL_Q + h * 8 + lane];
        }
        if (lane == 0) beta_s = betab[bt * HH + h];
        __syncthreads();

        // tmp = O @ c
        float s0 = 0.f, s1 = 0.f;
        #pragma unroll
        for (int k = 0; k < 8; ++k) {
            s0 = fmaf(O_s[i0 * 8 + k], c_s[k * 8 + j0], s0);
            s1 = fmaf(O_s[i1 * 8 + k], c_s[k * 8 + j1], s1);
        }
        t_s[e0] = s0; t_s[e1] = s1;
        __syncthreads();

        // r = tmp @ O^T
        s0 = 0.f; s1 = 0.f;
        #pragma unroll
        for (int k = 0; k < 8; ++k) {
            s0 = fmaf(t_s[i0 * 8 + k], O_s[j0 * 8 + k], s0);
            s1 = fmaf(t_s[i1 * 8 + k], O_s[j1 * 8 + k], s1);
        }
        r_s[e0] = s0; r_s[e1] = s1;
        __syncthreads();

        // kc = k^T @ r
        if (lane < 8) {
            float s = 0.f;
            #pragma unroll
            for (int i = 0; i < 8; ++i) s = fmaf(k_s[i], r_s[i * 8 + lane], s);
            kc_s[lane] = s;
        }
        __syncthreads();

        // c = r + beta * k (v - kc)^T
        const float bta = beta_s;
        c_s[e0] = fmaf(bta * k_s[i0], v_s[j0] - kc_s[j0], r_s[e0]);
        c_s[e1] = fmaf(bta * k_s[i1], v_s[j1] - kc_s[j1], r_s[e1]);
        __syncthreads();

        // o = q^T @ c
        if (lane < 8) {
            float s = 0.f;
            #pragma unroll
            for (int i = 0; i < 8; ++i) s = fmaf(q_s[i], c_s[i * 8 + lane], s);
            o_bf[bt * 128 + h * 8 + lane] = (__bf16)s;
        }
        __syncthreads();
    }
}

// ---------------------------------------------------------------------------
// Host-side launch
// ---------------------------------------------------------------------------
extern "C" void kernel_launch(void* const* d_in, const int* in_sizes, int n_in,
                              void* d_out, int out_size, void* d_ws, size_t ws_size,
                              hipStream_t stream)
{
    const float* x      = (const float*)d_in[0];
    const float* W_skew = (const float*)d_in[1];
    const float* W_k    = (const float*)d_in[2];
    const float* W_v    = (const float*)d_in[3];
    const float* W_q    = (const float*)d_in[4];
    const float* W_beta = (const float*)d_in[5];
    const float* b_beta = (const float*)d_in[6];
    const float* W_o    = (const float*)d_in[7];
    float* out = (float*)d_out;

    // workspace layout (all sizes 256B-aligned)
    char* w = (char*)d_ws;
    __bf16* xb    = (__bf16*)w;  w += (size_t)BT * DD * 2;        // 33.6 MB
    __bf16* Wt    = (__bf16*)w;  w += (size_t)PCOLS * DD * 2;     //  1.8 MB
    __bf16* Wot   = (__bf16*)w;  w += (size_t)1024 * 128 * 2;     //  0.3 MB
    float*  proj  = (float*)w;   w += (size_t)BT * PCOLS * 4;     // 58.7 MB
    float*  Obuf  = (float*)w;   w += (size_t)BT * HH * 64 * 4;   // 67.1 MB
    float*  knorm = (float*)w;   w += (size_t)BT * 128 * 4;       //  8.4 MB
    float*  betab = (float*)w;   w += (size_t)BT * HH * 4;        //  1.0 MB
    __bf16* o_bf  = (__bf16*)w;  w += (size_t)BT * 128 * 2;       //  4.2 MB

    // 1. x -> bf16
    {
        size_t n4 = (size_t)BT * DD / 4;
        cvt_f32_bf16_x4<<<(unsigned)((n4 + 255) / 256), 256, 0, stream>>>(x, xb, n4);
    }
    // 2. pack weights
    pack_weights<<<(PCOLS * DD + 255) / 256, 256, 0, stream>>>(
        W_skew, W_k, W_v, W_q, W_beta, Wt);
    pack_wo<<<(1024 * 128 + 255) / 256, 256, 0, stream>>>(W_o, Wot);

    // 3. projection GEMM: [16384 x 1024] @ [1024 x 896(padded)]
    {
        dim3 grid(PCOLS / (16 * NT), BT / (16 * 8));  // (14, 128)
        gemm_bf16_wmma<<<grid, 256, 0, stream>>>(xb, Wt, proj, BT, PCOLS, DD);
    }

    // 4. expm of 8x8 skews
    expm8_kernel<<<(BT * HH + 127) / 128, 128, 0, stream>>>(proj, Obuf, BT * HH);

    // 5. k-normalize + beta
    normbeta_kernel<<<(BT * HH + 255) / 256, 256, 0, stream>>>(
        proj, b_beta, knorm, betab, BT * HH);

    // 6. sequential scan: 64 chains, one wave32 each
    scan_kernel<<<BB * HH, 32, 0, stream>>>(Obuf, knorm, proj, betab, o_bf);

    // 7. output GEMM: [16384 x 128] @ [128 x 1024] -> d_out
    {
        dim3 grid(1024 / (16 * NT), BT / (16 * 8));   // (16, 128)
        gemm_bf16_wmma<<<grid, 256, 0, stream>>>(o_bf, Wot, out, BT, 1024, 128);
    }
}
